// Atten_919123001721
// MI455X (gfx1250) — matrix operand order repaired and checked by reading, not compiled, not run on hardware
//
#include <hip/hip_runtime.h>

typedef __attribute__((ext_vector_type(16))) _Float16 v16h;
typedef __attribute__((ext_vector_type(8)))  float    v8f;

#define NB 512
#define NS 77
#define NH 12
#define ND 64
#define NE 768
#define SP 80   // S padded to multiple of 16

// ---- WMMA helpers (gfx1250, wave32) --------------------------------------

static __device__ __forceinline__ v8f wmma_f16(v16h a, v16h b, v8f c) {
  return __builtin_amdgcn_wmma_f32_16x16x32_f16(
      false, a, false, b, (short)0, c, false, false);
}

// A-matrix 16x32 f16 fragment from row-major [m][k] buffer (global or LDS).
static __device__ __forceinline__ v16h load_a_f16(const _Float16* base, int ld) {
  const int lane = threadIdx.x & 31;
  const int m = lane & 15, half = lane >> 4;
  const _Float16* p = base + m * ld + half * 8;
  v16h f;
#pragma unroll
  for (int v = 0; v < 4; ++v) { f[2*v]   = p[2*v];      f[2*v+1]   = p[2*v+1]; }
#pragma unroll
  for (int v = 0; v < 4; ++v) { f[8+2*v] = p[16+2*v];   f[8+2*v+1] = p[16+2*v+1]; }
  return f;
}

// B-matrix 32x16 f16 fragment from buf[n][k] (B[k][n] = buf[n][k]).
static __device__ __forceinline__ v16h load_b_f16(const _Float16* base, int ld) {
  const int lane = threadIdx.x & 31;
  const int n = lane & 15, half = lane >> 4;
  const _Float16* p = base + n * ld + half * 8;
  v16h f;
#pragma unroll
  for (int v = 0; v < 8; ++v) { f[v] = p[v]; f[8+v] = p[16+v]; }
  return f;
}

// ---- CDNA5 async global->LDS copy (ASYNCcnt-tracked DMA) -----------------

static __device__ __forceinline__ uint32_t lds_byte_off(const void* p) {
  // generic pointer to __shared__: low 32 bits are the LDS address
  return (uint32_t)(uintptr_t)p;
}
static __device__ __forceinline__ void async_g2l_b128(uint32_t ldsoff, const void* g) {
  asm volatile("global_load_async_to_lds_b128 %0, %1, off"
               :: "v"(ldsoff), "v"((unsigned long long)(uintptr_t)g) : "memory");
}
static __device__ __forceinline__ void wait_async0() {
  asm volatile("s_wait_asynccnt 0x0" ::: "memory");
}

// ---- Kernel 0: convert + transpose weights to f16 [n][k] -----------------

__global__ void k_convert_w(const float* __restrict__ wq, const float* __restrict__ wk,
                            const float* __restrict__ wv, const float* __restrict__ wo,
                            _Float16* __restrict__ wT) {
  int i = blockIdx.x * 256 + threadIdx.x;        // exact grid: 4*E*E/256
  int mat = i / (NE * NE);
  int rem = i - mat * NE * NE;
  int n = rem / NE, k = rem - n * NE;
  const float* src = (mat == 0) ? wq : (mat == 1) ? wk : (mat == 2) ? wv : wo;
  wT[i] = (_Float16)src[k * NE + n];
}

// ---- Kernel 1: fused QKV projection --------------------------------------
// grid.x = NB (one batch element per block, all 80 padded rows).
// 256 threads = 8 waves; wave owns 288 cols of [Q|K|V] (2304), processed as
// 6 groups of 3 N-tiles; 5 M-tiles share each B load (acc[5][3]).

__global__ void __launch_bounds__(256) k_qkv(
    const float* __restrict__ x, const _Float16* __restrict__ wT,
    const float* __restrict__ bq, const float* __restrict__ bk,
    const float* __restrict__ bv,
    _Float16* __restrict__ qw, _Float16* __restrict__ kw, _Float16* __restrict__ vw) {
  __shared__ __align__(16) _Float16 Xs[SP * NE];   // 120 KB f16 x tile
  const int tid = threadIdx.x;
  const int b = blockIdx.x;

  // Stage x (80 rows, pad rows clamp to s=76), packed dword stores.
  for (int i = tid; i < SP * NE / 2; i += 256) {
    int r = i / (NE / 2), kp = i - r * (NE / 2);
    int s = r > NS - 1 ? NS - 1 : r;
    const float* xs = &x[((size_t)b * NS + s) * NE + kp * 2];
    union { _Float16 h[2]; uint32_t u; } pk;
    pk.h[0] = (_Float16)xs[0]; pk.h[1] = (_Float16)xs[1];
    ((uint32_t*)Xs)[i] = pk.u;
  }
  __syncthreads();

  const int wave = tid >> 5;
  const int lane15 = tid & 15, half = (tid >> 4) & 1;
  const v8f vzero = {0.f,0.f,0.f,0.f,0.f,0.f,0.f,0.f};

  for (int grp = 0; grp < 6; ++grp) {
    const int colBase = wave * 288 + grp * 48;
    v8f acc[5][3];
#pragma unroll
    for (int m = 0; m < 5; ++m)
#pragma unroll
      for (int t = 0; t < 3; ++t) acc[m][t] = vzero;

    for (int k0 = 0; k0 < NE; k0 += 32) {
      v16h a[5];
#pragma unroll
      for (int m = 0; m < 5; ++m) a[m] = load_a_f16(&Xs[m * 16 * NE + k0], NE);
#pragma unroll
      for (int t = 0; t < 3; ++t) {
        int c = colBase + t * 16;
        int mat = c / NE;
        int cc = c - mat * NE;
        v16h bf = load_b_f16(wT + (size_t)mat * NE * NE + (size_t)cc * NE + k0, NE);
#pragma unroll
        for (int m = 0; m < 5; ++m) acc[m][t] = wmma_f16(a[m], bf, acc[m][t]);
      }
    }
    // Epilogue: bias (+0.125 scale for Q), store f16 head-split [bH][80][64].
#pragma unroll
    for (int t = 0; t < 3; ++t) {
      int c = colBase + t * 16;
      int mat = c / NE;
      int cc = c - mat * NE;
      int col = cc + lane15;
      const float* bias = (mat == 0) ? bq : (mat == 1) ? bk : bv;
      _Float16* dst = (mat == 0) ? qw : (mat == 1) ? kw : vw;
      float bb = bias[col];
      float scale = (mat == 0) ? 0.125f : 1.0f;
      int h = col >> 6;
      int d = col & 63;
      size_t rb = (size_t)(b * NH + h) * SP * ND + d;
#pragma unroll
      for (int m = 0; m < 5; ++m) {
#pragma unroll
        for (int v = 0; v < 8; ++v) {
          int srow = m * 16 + v + 8 * half;      // 0..79 (pads stored, finite)
          dst[rb + (size_t)srow * ND] = (_Float16)((acc[m][t][v] + bb) * scale);
        }
      }
    }
  }
}

// ---- Kernel 2: causal attention per (b,h) --------------------------------
// grid.x = NB*NH; 160 threads = 5 waves; wave w handles q rows [16w,16w+16).

__global__ void __launch_bounds__(160) k_attn(
    const _Float16* __restrict__ qw, const _Float16* __restrict__ kw,
    const _Float16* __restrict__ vw, _Float16* __restrict__ ao) {
  __shared__ __align__(16) _Float16 Kt[SP * ND];   // K  [s][d]           10 KB
  __shared__ __align__(16) _Float16 Vt[ND * 96];   // V^T [d][s] pad->96  12 KB
  __shared__ __align__(16) _Float16 Pb[5][16 * 96];// per-wave probs      15 KB
  const int tid = threadIdx.x;
  const int bh = blockIdx.x;
  const int b = bh / NH, h = bh - b * NH;
  const size_t base = (size_t)bh * SP * ND;

  // stage K via async DMA (b128 granules: 80*64*2B / 16 = 640)
  {
    const uint32_t kt0 = lds_byte_off(&Kt[0]);
    const _Float16* src = kw + base;
    for (int i = tid; i < SP * ND / 8; i += 160)
      async_g2l_b128(kt0 + i * 16, src + i * 8);
  }
  // stage V transposed, zero-pad cols 80..95 (needs transpose -> VALU path)
  for (int i = tid; i < ND * 96; i += 160) {
    int d = i / 96, s = i - d * 96;
    Vt[i] = (s < SP) ? vw[base + (size_t)s * ND + d] : (_Float16)0.0f;
  }
  wait_async0();
  __syncthreads();

  const int wave = tid >> 5;
  const int lane15 = tid & 15, half = (tid >> 4) & 1;
  const v8f vzero = {0.f,0.f,0.f,0.f,0.f,0.f,0.f,0.f};

  // scores = Q K^T  (Q pre-scaled by 1/8)
  const _Float16* qbase = qw + base + (size_t)wave * 16 * ND;
  v16h aq0 = load_a_f16(qbase, ND);
  v16h aq1 = load_a_f16(qbase + 32, ND);
  v8f sc[5];
#pragma unroll
  for (int t = 0; t < 5; ++t) {
    v8f c = vzero;
    c = wmma_f16(aq0, load_b_f16(&Kt[t * 16 * ND], ND), c);
    c = wmma_f16(aq1, load_b_f16(&Kt[t * 16 * ND + 32], ND), c);
    sc[t] = c;
  }

  // causal softmax: row r = v + 8*half lives in one 16-lane half.
#pragma unroll
  for (int v = 0; v < 8; ++v) {
    int qrow = wave * 16 + v + 8 * half;
    float mx = -__builtin_inff();
#pragma unroll
    for (int t = 0; t < 5; ++t) {
      int col = t * 16 + lane15;
      float s = sc[t][v];
      if (col > qrow) s = -__builtin_inff();   // also kills pad cols 77..79
      sc[t][v] = s;
      mx = fmaxf(mx, s);
    }
#pragma unroll
    for (int o = 1; o < 16; o <<= 1) mx = fmaxf(mx, __shfl_xor(mx, o, 32));
    float sum = 0.f;
#pragma unroll
    for (int t = 0; t < 5; ++t) { float p = __expf(sc[t][v] - mx); sc[t][v] = p; sum += p; }
#pragma unroll
    for (int o = 1; o < 16; o <<= 1) sum += __shfl_xor(sum, o, 32);
    float inv = 1.0f / sum;
#pragma unroll
    for (int t = 0; t < 5; ++t) sc[t][v] *= inv;
  }

  // round-trip probs through LDS to re-fragment as A (16x96, zero-padded)
  _Float16* Pw = &Pb[wave][0];
#pragma unroll
  for (int t = 0; t < 5; ++t)
#pragma unroll
    for (int v = 0; v < 8; ++v) {
      int m = v + 8 * half;
      Pw[m * 96 + t * 16 + lane15] = (_Float16)sc[t][v];
    }
#pragma unroll
  for (int v = 0; v < 8; ++v) {
    int m = v + 8 * half;
    Pw[m * 96 + 80 + lane15] = (_Float16)0.0f;
  }

  // out = P @ V   (K-dim 96 = 3 x 32, padded region contributes exact 0)
  v16h ap0 = load_a_f16(Pw + 0,  96);
  v16h ap1 = load_a_f16(Pw + 32, 96);
  v16h ap2 = load_a_f16(Pw + 64, 96);
#pragma unroll
  for (int nt = 0; nt < 4; ++nt) {
    v8f o = vzero;
    o = wmma_f16(ap0, load_b_f16(&Vt[nt * 16 * 96 + 0],  96), o);
    o = wmma_f16(ap1, load_b_f16(&Vt[nt * 16 * 96 + 32], 96), o);
    o = wmma_f16(ap2, load_b_f16(&Vt[nt * 16 * 96 + 64], 96), o);
#pragma unroll
    for (int v = 0; v < 8; ++v) {
      int m = v + 8 * half;
      int s = wave * 16 + m;
      if (s < NS)
        ao[((size_t)b * NS + s) * NE + h * ND + nt * 16 + lane15] = (_Float16)o[v];
    }
  }
}

// ---- Kernel 3: output projection + bias, fp32 out ------------------------
// grid.x = 39424/64 = 616 blocks of 64 rows; 256 threads = 8 waves.
// 4 M-tiles share each B load (acc[4][3]); A staged via async DMA (96 KB).

__global__ void __launch_bounds__(256) k_oproj(
    const _Float16* __restrict__ ao, const _Float16* __restrict__ woT,
    const float* __restrict__ bo, float* __restrict__ out) {
  __shared__ __align__(16) _Float16 As[64 * NE];   // 96 KB
  const int tid = threadIdx.x;
  const int mb = blockIdx.x;                       // 64-row block
  {
    const uint32_t as0 = lds_byte_off(&As[0]);
    const _Float16* src = ao + (size_t)mb * 64 * NE;
    for (int i = tid; i < 64 * NE / 8; i += 256)   // 6144 b128 granules
      async_g2l_b128(as0 + i * 16, src + i * 8);
  }
  wait_async0();
  __syncthreads();

  const int wave = tid >> 5;
  const int lane15 = tid & 15, half = (tid >> 4) & 1;
  const v8f vzero = {0.f,0.f,0.f,0.f,0.f,0.f,0.f,0.f};

  for (int grp = 0; grp < 2; ++grp) {
    const int colBase = wave * 96 + grp * 48;
    v8f acc[4][3];
#pragma unroll
    for (int m = 0; m < 4; ++m)
#pragma unroll
      for (int t = 0; t < 3; ++t) acc[m][t] = vzero;

    for (int k0 = 0; k0 < NE; k0 += 32) {
      v16h a[4];
#pragma unroll
      for (int m = 0; m < 4; ++m) a[m] = load_a_f16(&As[m * 16 * NE + k0], NE);
#pragma unroll
      for (int t = 0; t < 3; ++t) {
        v16h bf = load_b_f16(woT + (size_t)(colBase + t * 16) * NE + k0, NE);
#pragma unroll
        for (int m = 0; m < 4; ++m) acc[m][t] = wmma_f16(a[m], bf, acc[m][t]);
      }
    }
#pragma unroll
    for (int t = 0; t < 3; ++t) {
      int c = colBase + t * 16 + lane15;
      float bb = bo[c];
#pragma unroll
      for (int m = 0; m < 4; ++m) {
#pragma unroll
        for (int v = 0; v < 8; ++v) {
          int row = mb * 64 + m * 16 + v + 8 * half;
          out[(size_t)row * NE + c] = acc[m][t][v] + bb;
        }
      }
    }
  }
}

// ---- host launch ---------------------------------------------------------

extern "C" void kernel_launch(void* const* d_in, const int* in_sizes, int n_in,
                              void* d_out, int out_size, void* d_ws, size_t ws_size,
                              hipStream_t stream) {
  (void)in_sizes; (void)n_in; (void)out_size; (void)ws_size;
  const float* x  = (const float*)d_in[0];
  const float* wq = (const float*)d_in[1];
  const float* bq = (const float*)d_in[2];
  const float* wk = (const float*)d_in[3];
  const float* bk = (const float*)d_in[4];
  const float* wv = (const float*)d_in[5];
  const float* bv = (const float*)d_in[6];
  const float* wo = (const float*)d_in[7];
  const float* bo = (const float*)d_in[8];
  float* out = (float*)d_out;

  // workspace (f16): wT[4*E*E] | Q | K | V (each B*H*80*64) | AO (B*S*E)
  _Float16* wT = (_Float16*)d_ws;
  size_t off = (size_t)4 * NE * NE;
  const size_t bhsd = (size_t)NB * NH * SP * ND;
  _Float16* qws = wT + off;          off += bhsd;
  _Float16* kws = wT + off;          off += bhsd;
  _Float16* vws = wT + off;          off += bhsd;
  _Float16* aob = wT + off;

  k_convert_w<<<4 * NE * NE / 256, 256, 0, stream>>>(wq, wk, wv, wo, wT);
  k_qkv<<<NB, 256, 0, stream>>>(x, wT, bq, bk, bv, qws, kws, vws);
  k_attn<<<NB * NH, 160, 0, stream>>>(qws, kws, vws, aob);
  k_oproj<<<(NB * NS) / 64, 256, 0, stream>>>(aob, wT + (size_t)3 * NE * NE, bo, out);
}
